// GroupedQueryAttention_9131100471592
// MI455X (gfx1250) — compile-verified, matrix operand order checked by reading, not run on hardware
//
#include <hip/hip_runtime.h>

typedef unsigned short u16;
typedef unsigned long long u64;
typedef __attribute__((ext_vector_type(16))) __bf16 v16bf;
typedef __attribute__((ext_vector_type(8)))  float  v8f;

#define B_  2
#define S_  2048
#define D_  2048
#define H_  32
#define KV_ 8
#define G_  4
#define DK_ 64
#define M_  (B_*S_)   // 4096 rows

static __device__ inline u16 f2bfu(float f) {
  __bf16 h = (__bf16)f;                       // RNE convert
  return __builtin_bit_cast(u16, h);
}
static __device__ inline __bf16 u2bf(u16 u) { return __builtin_bit_cast(__bf16, u); }
static __device__ inline float  bfu2f(u16 u) { return (float)__builtin_bit_cast(__bf16, u); }

// gfx1250 async copy: global -> LDS, 16B per lane, tracked by ASYNCcnt.
// lds_off = low 32 bits of generic LDS pointer.
static __device__ inline void async_ld16(unsigned lds_off, const void* g) {
  asm volatile("global_load_async_to_lds_b128 %0, %1, off"
               :: "v"(lds_off), "v"((u64)(uintptr_t)g) : "memory");
}
#define WAIT_ASYNC_4() asm volatile("s_wait_asynccnt 0x4" ::: "memory")
#define WAIT_ASYNC_0() asm volatile("s_wait_asynccnt 0x0" ::: "memory")

// ---------------------------------------------------------------- fp32 -> bf16
__global__ void cvt_f32_bf16(const float* __restrict__ in, u16* __restrict__ out, int n) {
  int i = blockIdx.x * blockDim.x + threadIdx.x;
  int stride = gridDim.x * blockDim.x;
  for (; i < n; i += stride) out[i] = f2bfu(in[i]);
}

// ---------------------------------------------------------------- GEMM: Y = X * W^T
// X: [M,K] bf16 row-major, W: [N,K] bf16 row-major (torch Linear layout), Y: [M,N] fp32.
// Block 256 threads (8 waves), tile 128x128, K-step 32, double-buffered async LDS fill.
__global__ __launch_bounds__(256)
void gemm_xwT(const u16* __restrict__ X, const u16* __restrict__ W,
              float* __restrict__ Y, int M, int N, int K) {
  // stride 40 elems = 80B: 16B-aligned rows + conflict-light bank spread
  __shared__ u16 Xs[2][128][40];
  __shared__ u16 Ws[2][128][40];
  const int m0 = blockIdx.y * 128;
  const int n0 = blockIdx.x * 128;
  const int wave = threadIdx.x >> 5;
  const int lane = threadIdx.x & 31;
  const int l15  = lane & 15;
  const int hi   = lane >> 4;
  const int kb8  = hi * 8;
  const int wm = wave >> 1;      // 0..3 : 32-row slab
  const int wn = wave & 1;       // 0..1 : 64-col slab

  const unsigned xsBase = (unsigned)(uintptr_t)&Xs[0][0][0];
  const unsigned wsBase = (unsigned)(uintptr_t)&Ws[0][0][0];
  const unsigned bufStride = 128u * 40u * 2u;   // bytes per LDS buffer

  // per-thread load slots: 512 b128 units per 128x32 tile, 2 per thread per matrix
  const int ua = threadIdx.x, ub = threadIdx.x + 256;
  const int ra = ua >> 2, ca = (ua & 3) * 8;
  const int rb = ub >> 2, cb = (ub & 3) * 8;

  auto issue = [&](int k0, int buf) {
    unsigned xo = xsBase + buf * bufStride;
    unsigned wo = wsBase + buf * bufStride;
    async_ld16(xo + (unsigned)(ra * 40 + ca) * 2, X + (size_t)(m0 + ra) * K + k0 + ca);
    async_ld16(xo + (unsigned)(rb * 40 + cb) * 2, X + (size_t)(m0 + rb) * K + k0 + cb);
    async_ld16(wo + (unsigned)(ra * 40 + ca) * 2, W + (size_t)(n0 + ra) * K + k0 + ca);
    async_ld16(wo + (unsigned)(rb * 40 + cb) * 2, W + (size_t)(n0 + rb) * K + k0 + cb);
  };

  v8f acc[2][4];
  #pragma unroll
  for (int a = 0; a < 2; ++a)
    #pragma unroll
    for (int b = 0; b < 4; ++b)
      acc[a][b] = (v8f){0.f,0.f,0.f,0.f,0.f,0.f,0.f,0.f};

  const int nk = K / 32;
  issue(0, 0);
  for (int kt = 0; kt < nk; ++kt) {
    const int buf = kt & 1;
    if (kt + 1 < nk) {
      issue((kt + 1) * 32, buf ^ 1);
      WAIT_ASYNC_4();                 // first 4 outstanding (current tile) are done
    } else {
      WAIT_ASYNC_0();
    }
    __syncthreads();                  // current tile visible to all waves

    // A fragments: 16-bit A 16x32 layout (lane<16: K 0-7 & 16-23; lane>=16: K 8-15 & 24-31)
    v16bf afrag[2];
    #pragma unroll
    for (int ms = 0; ms < 2; ++ms) {
      const u16* xr = &Xs[buf][wm * 32 + ms * 16 + l15][0];
      #pragma unroll
      for (int i = 0; i < 8; ++i) {
        afrag[ms][i]     = u2bf(xr[kb8 + i]);
        afrag[ms][8 + i] = u2bf(xr[16 + kb8 + i]);
      }
    }
    // B fragments: B[k][n] = W[n][k]; lane holds column n = l15
    #pragma unroll
    for (int j = 0; j < 4; ++j) {
      v16bf bfrag;
      const u16* wr = &Ws[buf][wn * 64 + j * 16 + l15][0];
      #pragma unroll
      for (int i = 0; i < 8; ++i) {
        bfrag[i]     = u2bf(wr[kb8 + i]);
        bfrag[8 + i] = u2bf(wr[16 + kb8 + i]);
      }
      #pragma unroll
      for (int ms = 0; ms < 2; ++ms)
        acc[ms][j] = __builtin_amdgcn_wmma_f32_16x16x32_bf16(
            false, afrag[ms], false, bfrag, (short)0, acc[ms][j], false, false);
    }
    __syncthreads();                  // all reads done before buffer is refilled
  }

  // C/D layout: VGPR i -> row i + 8*hi, col = l15
  #pragma unroll
  for (int ms = 0; ms < 2; ++ms)
    #pragma unroll
    for (int j = 0; j < 4; ++j)
      #pragma unroll
      for (int i = 0; i < 8; ++i) {
        int row = m0 + wm * 32 + ms * 16 + i + 8 * hi;
        int col = n0 + wn * 64 + j * 16 + l15;
        Y[(size_t)row * N + col] = acc[ms][j][i];
      }
}

// ---------------------------------------------------------------- RoPE + [b,s,h,d]->[b,h,s,d] + bf16
// Interleaved RoPE: out[2j] = e*cos - o*sin ; out[2j+1] = o*cos + e*sin
__global__ void rope_pack(const float* __restrict__ Y, const float* __restrict__ invf,
                          const int* __restrict__ pos_ids, u16* __restrict__ out, int Hx) {
  int idx = blockIdx.x * blockDim.x + threadIdx.x;
  int total = B_ * S_ * Hx * (DK_ / 2);
  if (idx >= total) return;
  int j  = idx & (DK_ / 2 - 1);            // rotation pair 0..31
  int h  = (idx / (DK_ / 2)) % Hx;
  int bs = idx / ((DK_ / 2) * Hx);
  int s  = bs & (S_ - 1);
  int b  = bs / S_;
  float p = (float)pos_ids[bs];
  float ang = p * invf[j];
  float sn, cs;
  __sincosf(ang, &sn, &cs);
  const float* src = Y + ((size_t)bs * Hx + h) * DK_ + 2 * j;
  float e = src[0], o = src[1];
  u16* dst = out + (((size_t)(b * Hx + h) * S_ + s) * DK_) + 2 * j;
  dst[0] = f2bfu(e * cs - o * sn);
  dst[1] = f2bfu(o * cs + e * sn);
}

// ---------------------------------------------------------------- V: [b,s,kv,d] -> [b,kv,s,d] bf16
__global__ void v_pack(const float* __restrict__ Y, u16* __restrict__ out) {
  int idx = blockIdx.x * blockDim.x + threadIdx.x;
  int total = B_ * S_ * KV_ * DK_;
  if (idx >= total) return;
  int d  = idx & (DK_ - 1);
  int kv = (idx / DK_) % KV_;
  int bs = idx / (DK_ * KV_);
  int s  = bs & (S_ - 1);
  int b  = bs / S_;
  out[(((size_t)(b * KV_ + kv) * S_ + s) * DK_) + d] = f2bfu(Y[idx]);
}

// ---------------------------------------------------------------- flash attention
// grid.x = B*H*(S/64); block = 128 (4 waves, 16 queries each).
// Computes S^T = K*Q^T so the exp'd C-fragment is directly the B operand of ctx^T = V^T*P^T.
// K/V tiles double-buffered in LDS via async global->LDS copies.
__global__ __launch_bounds__(128)
void flash_attn(const u16* __restrict__ Q, const u16* __restrict__ Kb,
                const u16* __restrict__ Vb, u16* __restrict__ ctx) {
  __shared__ u16 Ks[2][32][72];    // stride 72 elems = 144B: 16B-aligned, conflict-free rows
  __shared__ u16 Vs[2][32][72];
  const int qtiles = S_ / 64;
  const int qt   = blockIdx.x % qtiles;
  const int head = blockIdx.x / qtiles;        // b*H + h
  const int b  = head / H_;
  const int h  = head % H_;
  const int kv = h / G_;                        // GQA: query head h -> kv head h/G
  const u16* Qh = Q  + ((size_t)head * S_) * DK_;
  const u16* Kh = Kb + ((size_t)(b * KV_ + kv) * S_) * DK_;
  const u16* Vh = Vb + ((size_t)(b * KV_ + kv) * S_) * DK_;

  const int wave = threadIdx.x >> 5;
  const int lane = threadIdx.x & 31;
  const int l15  = lane & 15;
  const int hi   = lane >> 4;
  const int kb8  = hi * 8;
  const int q0   = qt * 64 + wave * 16;
  const float LOG2E = 1.4426950408889634f;

  const unsigned ksBase = (unsigned)(uintptr_t)&Ks[0][0][0];
  const unsigned vsBase = (unsigned)(uintptr_t)&Vs[0][0][0];
  const unsigned bufStride = 32u * 72u * 2u;

  // per-thread load slots: 256 b128 units per 32x64 tile, 2 per thread per matrix
  const int ua = threadIdx.x, ub = threadIdx.x + 128;
  const int ra = ua >> 3, ca = (ua & 7) * 8;
  const int rb = ub >> 3, cb = (ub & 7) * 8;

  auto issue = [&](int k0, int buf) {
    unsigned ko = ksBase + buf * bufStride;
    unsigned vo = vsBase + buf * bufStride;
    async_ld16(ko + (unsigned)(ra * 72 + ca) * 2, Kh + (size_t)(k0 + ra) * DK_ + ca);
    async_ld16(ko + (unsigned)(rb * 72 + cb) * 2, Kh + (size_t)(k0 + rb) * DK_ + cb);
    async_ld16(vo + (unsigned)(ra * 72 + ca) * 2, Vh + (size_t)(k0 + ra) * DK_ + ca);
    async_ld16(vo + (unsigned)(rb * 72 + cb) * 2, Vh + (size_t)(k0 + rb) * DK_ + cb);
  };

  // Q^T B-fragments (feat chunks 0-31, 32-63), pre-scaled by 1/sqrt(64)=0.125 (exact in bf16)
  v16bf qf[2];
  #pragma unroll
  for (int c = 0; c < 2; ++c) {
    const u16* qr = Qh + (size_t)(q0 + l15) * DK_ + c * 32;
    #pragma unroll
    for (int i = 0; i < 8; ++i) {
      qf[c][i]     = (__bf16)(bfu2f(qr[kb8 + i])      * 0.125f);
      qf[c][8 + i] = (__bf16)(bfu2f(qr[16 + kb8 + i]) * 0.125f);
    }
  }

  v8f cacc[4];
  #pragma unroll
  for (int t = 0; t < 4; ++t) cacc[t] = (v8f){0.f,0.f,0.f,0.f,0.f,0.f,0.f,0.f};
  float mrun = -1e30f, lrun = 0.f;

  const int nk = S_ / 32;
  issue(0, 0);
  for (int kt = 0; kt < nk; ++kt) {
    const int buf = kt & 1;
    if (kt + 1 < nk) {
      issue((kt + 1) * 32, buf ^ 1);
      WAIT_ASYNC_4();
    } else {
      WAIT_ASYNC_0();
    }
    __syncthreads();

    // S^T[key][q] for two 16-key sub-tiles, accumulating over the 64-feat dim
    v8f sacc[2];
    sacc[0] = (v8f){0.f,0.f,0.f,0.f,0.f,0.f,0.f,0.f};
    sacc[1] = (v8f){0.f,0.f,0.f,0.f,0.f,0.f,0.f,0.f};
    #pragma unroll
    for (int n = 0; n < 2; ++n) {
      #pragma unroll
      for (int c = 0; c < 2; ++c) {
        v16bf kf;
        const u16* kr = &Ks[buf][n * 16 + l15][c * 32];
        #pragma unroll
        for (int i = 0; i < 8; ++i) {
          kf[i]     = u2bf(kr[kb8 + i]);
          kf[8 + i] = u2bf(kr[16 + kb8 + i]);
        }
        sacc[n] = __builtin_amdgcn_wmma_f32_16x16x32_bf16(
            false, kf, false, qf[c], (short)0, sacc[n], false, false);
      }
    }

    // online softmax over this 32-key block; row (query) = l15, spread over lanes l & l+16
    float mx = -1e30f;
    #pragma unroll
    for (int i = 0; i < 8; ++i) {
      mx = fmaxf(mx, sacc[0][i]);
      mx = fmaxf(mx, sacc[1][i]);
    }
    mx = fmaxf(mx, __shfl_xor(mx, 16, 32));
    float mnew = fmaxf(mrun, mx);
    float corr = exp2f((mrun - mnew) * LOG2E);

    // P^T: lands exactly in the B-fragment layout of the next WMMA (keys kb8.., 16+kb8..)
    v16bf pf;
    float ps = 0.f;
    #pragma unroll
    for (int i = 0; i < 8; ++i) {
      float p0 = exp2f((sacc[0][i] - mnew) * LOG2E);   // keys  kb8+i
      float p1 = exp2f((sacc[1][i] - mnew) * LOG2E);   // keys 16+kb8+i
      ps += p0 + p1;
      pf[i]     = (__bf16)p0;
      pf[8 + i] = (__bf16)p1;
    }
    ps += __shfl_xor(ps, 16, 32);
    lrun = lrun * corr + ps;
    mrun = mnew;

    #pragma unroll
    for (int t = 0; t < 4; ++t)
      #pragma unroll
      for (int i = 0; i < 8; ++i) cacc[t][i] *= corr;

    // ctx^T += V^T * P^T  (A = V^T: lane = feat row, elements = keys)
    #pragma unroll
    for (int t = 0; t < 4; ++t) {
      v16bf vf;
      #pragma unroll
      for (int i = 0; i < 8; ++i) {
        vf[i]     = u2bf(Vs[buf][kb8 + i][t * 16 + l15]);
        vf[8 + i] = u2bf(Vs[buf][16 + kb8 + i][t * 16 + l15]);
      }
      cacc[t] = __builtin_amdgcn_wmma_f32_16x16x32_bf16(
          false, vf, false, pf, (short)0, cacc[t], false, false);
    }
    __syncthreads();                 // reads done before this buffer is refilled
  }

  // normalize and store ctx as [b, s, h, dk] == [M, D] bf16 for the O-projection GEMM
  float inv = 1.0f / lrun;
  #pragma unroll
  for (int t = 0; t < 4; ++t)
    #pragma unroll
    for (int i = 0; i < 8; ++i) {
      int f    = t * 16 + i + 8 * hi;       // ctx^T C layout: row=feat, col=query
      int srow = q0 + l15;
      ctx[(((size_t)(b * S_ + srow) * H_) + h) * DK_ + f] = f2bfu(cacc[t][i] * inv);
    }
}

// ----------------------------------------------------------------
extern "C" void kernel_launch(void* const* d_in, const int* in_sizes, int n_in,
                              void* d_out, int out_size, void* d_ws, size_t ws_size,
                              hipStream_t stream) {
  (void)in_sizes; (void)n_in; (void)out_size; (void)ws_size;
  const float* x    = (const float*)d_in[0];
  const int*   pos  = (const int*)d_in[1];
  const float* Wq   = (const float*)d_in[2];
  const float* Wk   = (const float*)d_in[3];
  const float* Wv   = (const float*)d_in[4];
  const float* Wo   = (const float*)d_in[5];
  const float* invf = (const float*)d_in[6];
  float* out = (float*)d_out;

  char* base = (char*)d_ws;
  size_t off = 0;
  auto carve = [&](size_t bytes) -> void* {
    void* p = base + off;
    off += (bytes + 255) & ~(size_t)255;
    return p;
  };
  u16*   xb  = (u16*)  carve((size_t)M_ * D_ * 2);
  u16*   Wqb = (u16*)  carve((size_t)D_ * D_ * 2);
  u16*   Wkb = (u16*)  carve((size_t)(KV_ * DK_) * D_ * 2);
  u16*   Wvb = (u16*)  carve((size_t)(KV_ * DK_) * D_ * 2);
  u16*   Wob = (u16*)  carve((size_t)D_ * D_ * 2);
  float* Yq  = (float*)carve((size_t)M_ * D_ * 4);
  float* Yk  = (float*)carve((size_t)M_ * (KV_ * DK_) * 4);
  float* Yv  = (float*)carve((size_t)M_ * (KV_ * DK_) * 4);
  u16*   Qb  = (u16*)  carve((size_t)M_ * D_ * 2);
  u16*   Kb  = (u16*)  carve((size_t)B_ * KV_ * S_ * DK_ * 2);
  u16*   Vb  = (u16*)  carve((size_t)B_ * KV_ * S_ * DK_ * 2);
  u16*   ctx = (u16*)Yq;   // Yq dead after rope_pack; reuse for attention output

  cvt_f32_bf16<<<dim3(1024), dim3(256), 0, stream>>>(x,  xb,  M_ * D_);
  cvt_f32_bf16<<<dim3(1024), dim3(256), 0, stream>>>(Wq, Wqb, D_ * D_);
  cvt_f32_bf16<<<dim3(256),  dim3(256), 0, stream>>>(Wk, Wkb, KV_ * DK_ * D_);
  cvt_f32_bf16<<<dim3(256),  dim3(256), 0, stream>>>(Wv, Wvb, KV_ * DK_ * D_);
  cvt_f32_bf16<<<dim3(1024), dim3(256), 0, stream>>>(Wo, Wob, D_ * D_);

  gemm_xwT<<<dim3(D_ / 128,           M_ / 128), dim3(256), 0, stream>>>(xb, Wqb, Yq, M_, D_,          D_);
  gemm_xwT<<<dim3((KV_ * DK_) / 128,  M_ / 128), dim3(256), 0, stream>>>(xb, Wkb, Yk, M_, KV_ * DK_,   D_);
  gemm_xwT<<<dim3((KV_ * DK_) / 128,  M_ / 128), dim3(256), 0, stream>>>(xb, Wvb, Yv, M_, KV_ * DK_,   D_);

  rope_pack<<<dim3((B_ * S_ * H_  * (DK_ / 2) + 255) / 256), dim3(256), 0, stream>>>(Yq, invf, pos, Qb, H_);
  rope_pack<<<dim3((B_ * S_ * KV_ * (DK_ / 2) + 255) / 256), dim3(256), 0, stream>>>(Yk, invf, pos, Kb, KV_);
  v_pack  <<<dim3((B_ * S_ * KV_ * DK_ + 255) / 256),        dim3(256), 0, stream>>>(Yv, Vb);

  flash_attn<<<dim3(B_ * H_ * (S_ / 64)), dim3(128), 0, stream>>>(Qb, Kb, Vb, ctx);

  gemm_xwT<<<dim3(D_ / 128, M_ / 128), dim3(256), 0, stream>>>(ctx, Wob, out, M_, D_, D_);
}
